// OTTT_67181878444196
// MI455X (gfx1250) — matrix operand order repaired and checked by reading, not compiled, not run on hardware
//
#include <hip/hip_runtime.h>
#include <hip/hip_bf16.h>
#include <stdint.h>

// ---------------------------------------------------------------------------
// OTTT spiking timestep for MI455X (gfx1250, wave32, WMMA).
//   I      = x @ W              (split-bf16 3-pass WMMA, f32 accumulate)
//   u_new  = sig_tau*u + I
//   s      = (u_new >= 1)
//   u_out  = u_new - s
//   a_hat' = sig_tau*a_hat + x
// Outputs concatenated: [u_out (B*OUT) | a_hat' (B*IN) | s (B*OUT)]
// ---------------------------------------------------------------------------

#define B_ROWS 16384
#define IN_F   512
#define OUT_F  512
// sigmoid(2.0)
#define SIG_TAU 0.88079707797788231f

typedef __attribute__((ext_vector_type(16))) __bf16 v16bf;
typedef __attribute__((ext_vector_type(8)))  float  v8f;

union BFrag {
    v16bf v;
    uint4 q[2];
};

__device__ __forceinline__ unsigned short bf16_rne(float f) {
    unsigned int u = __float_as_uint(f);
    unsigned int r = u + 0x7FFFu + ((u >> 16) & 1u);
    return (unsigned short)(r >> 16);
}

// ---------------------------------------------------------------------------
// Kernel 1: split W (f32, [K=512][N=512] row-major) into bf16 hi/lo, laid out
// as ready-to-load WMMA B fragments:
//   frag f = ntile*16 + kb   (ntile: N/16, kb: K/32)
//   bytes: ws + f*1024 + half*512 + lane*16       (hi)
//          ws + 512KB + same                       (lo)
// Lane n (<16): col=n,       K = kb*32 + j      (j = half*8 .. half*8+7 pairs)
// Lane n (>=16): col=n-16,   K = kb*32 + 16 + j
// ---------------------------------------------------------------------------
__global__ void ottt_prep_w(const float* __restrict__ W,
                            unsigned char* __restrict__ ws) {
    int t = blockIdx.x * blockDim.x + threadIdx.x;   // 16384 threads
    int lane  = t & 31;
    int kb    = (t >> 5) & 15;
    int ntile = t >> 9;                              // 0..31
    int col   = ntile * 16 + (lane & 15);
    int kbase = kb * 32 + ((lane >> 4) << 4);

    unsigned int hs[8], ls[8];
#pragma unroll
    for (int j = 0; j < 16; ++j) {
        float w = W[(size_t)(kbase + j) * OUT_F + col];
        unsigned short h = bf16_rne(w);
        float fh = __uint_as_float((unsigned int)h << 16);
        unsigned short l = bf16_rne(w - fh);
        if (j & 1) { hs[j >> 1] |= (unsigned int)h << 16; ls[j >> 1] |= (unsigned int)l << 16; }
        else       { hs[j >> 1]  = (unsigned int)h;       ls[j >> 1]  = (unsigned int)l;       }
    }
    int f = ntile * 16 + kb;
    uint4* hiP = (uint4*)(ws + (size_t)f * 1024 + lane * 16);
    uint4* loP = (uint4*)(ws + (size_t)(512 * 1024) + (size_t)f * 1024 + lane * 16);
    hiP[0]  = make_uint4(hs[0], hs[1], hs[2], hs[3]);   // half 0 (j=0..7)
    hiP[32] = make_uint4(hs[4], hs[5], hs[6], hs[7]);   // half 1 (j=8..15), +512B
    loP[0]  = make_uint4(ls[0], ls[1], ls[2], ls[3]);
    loP[32] = make_uint4(ls[4], ls[5], ls[6], ls[7]);
}

// ---------------------------------------------------------------------------
// Kernel 2: fused GEMM (split-bf16 WMMA) + LIF update.
// Grid: B/32 workgroups of 256 threads (8 waves).
// Each WG: 32 rows of x -> LDS (bf16 hi/lo, row-rotated to spread LDS banks).
// Wave w: M-tile (w&1), 8 N-tiles starting at (w>>1)*8.
// ---------------------------------------------------------------------------
__global__ __launch_bounds__(256) void ottt_gemm_lif(
        const float* __restrict__ u,
        const float* __restrict__ x,
        const unsigned char* __restrict__ wsw,
        float* __restrict__ u_out,
        float* __restrict__ s_out) {
    __shared__ unsigned short ldsH[32 * 512];   // 32KB
    __shared__ unsigned short ldsL[32 * 512];   // 32KB

    const int tid  = threadIdx.x;
    const int row0 = blockIdx.x * 32;

    // ---- Phase 1: load + split x rows into LDS (rotated by row*8 halfs) ----
#pragma unroll
    for (int it = 0; it < 16; ++it) {
        int flat = it * 1024 + tid * 4;          // 256 thr * 4 f32 per iter
        int row  = flat >> 9;
        int col  = flat & 511;
        const float4 xv = *(const float4*)(x + (size_t)(row0 + row) * IN_F + col);
        float e[4] = {xv.x, xv.y, xv.z, xv.w};
        unsigned short h[4], l[4];
#pragma unroll
        for (int j = 0; j < 4; ++j) {
            h[j] = bf16_rne(e[j]);
            float fh = __uint_as_float((unsigned int)h[j] << 16);
            l[j] = bf16_rne(e[j] - fh);
        }
        int colr = (col + row * 8) & 511;        // bank-spread rotation
        int idx  = row * 512 + colr;
        *(uint2*)&ldsH[idx] = make_uint2((unsigned)h[0] | ((unsigned)h[1] << 16),
                                         (unsigned)h[2] | ((unsigned)h[3] << 16));
        *(uint2*)&ldsL[idx] = make_uint2((unsigned)l[0] | ((unsigned)l[1] << 16),
                                         (unsigned)l[2] | ((unsigned)l[3] << 16));
    }
    __syncthreads();

    // ---- Phase 2: WMMA K-loop ----
    const int lane   = tid & 31;
    const int wid    = tid >> 5;
    const int mtile  = wid & 1;               // 0/1 -> rows [0..15]/[16..31]
    const int ntbase = (wid >> 1) * 8;        // first of 8 N-tiles
    const int arow   = mtile * 16 + (lane & 15);
    const int ksel   = (lane >> 4) * 16;      // lanes>=16 hold K+16 halves

    v8f acc[8] = {};

    for (int kb = 0; kb < 16; ++kb) {
        BFrag aH, aL;
        int k0 = kb * 32 + ksel;
        int c0 = (k0 + arow * 8) & 511;
        int c1 = ((k0 + 8) + arow * 8) & 511;
        aH.q[0] = *(const uint4*)&ldsH[arow * 512 + c0];
        aH.q[1] = *(const uint4*)&ldsH[arow * 512 + c1];
        aL.q[0] = *(const uint4*)&ldsL[arow * 512 + c0];
        aL.q[1] = *(const uint4*)&ldsL[arow * 512 + c1];

#pragma unroll
        for (int t = 0; t < 8; ++t) {
            int f = (ntbase + t) * 16 + kb;
            const uint4* bh = (const uint4*)(wsw + (size_t)f * 1024 + lane * 16);
            const uint4* bl = (const uint4*)(wsw + (size_t)(512 * 1024) +
                                             (size_t)f * 1024 + lane * 16);
            BFrag bH, bL;
            bH.q[0] = bh[0];  bH.q[1] = bh[32];
            bL.q[0] = bl[0];  bL.q[1] = bl[32];

            acc[t] = __builtin_amdgcn_wmma_f32_16x16x32_bf16(
                false, aH.v, false, bH.v, (short)0, acc[t], false, false);
            acc[t] = __builtin_amdgcn_wmma_f32_16x16x32_bf16(
                false, aH.v, false, bL.v, (short)0, acc[t], false, false);
            acc[t] = __builtin_amdgcn_wmma_f32_16x16x32_bf16(
                false, aL.v, false, bH.v, (short)0, acc[t], false, false);
        }
    }

    // ---- Phase 3: fused LIF epilogue directly from WMMA C/D layout ----
    // acc element r, this lane -> (M = r + 8*(lane>=16), N = lane&15) in tile.
    const int rbase = row0 + mtile * 16 + (lane >> 4) * 8;
#pragma unroll
    for (int t = 0; t < 8; ++t) {
        int col = (ntbase + t) * 16 + (lane & 15);
#pragma unroll
        for (int r = 0; r < 8; ++r) {
            size_t idx = (size_t)(rbase + r) * OUT_F + col;
            float un = SIG_TAU * u[idx] + acc[t][r];
            float s  = (un >= 1.0f) ? 1.0f : 0.0f;
            u_out[idx] = un - s;
            s_out[idx] = s;
        }
    }
}

// ---------------------------------------------------------------------------
// Fallback GEMM+LIF (no d_ws): scalar f32, only if ws_size < 1MB.
// ---------------------------------------------------------------------------
__global__ void ottt_gemm_lif_fallback(const float* __restrict__ u,
                                       const float* __restrict__ x,
                                       const float* __restrict__ W,
                                       float* __restrict__ u_out,
                                       float* __restrict__ s_out) {
    size_t i = (size_t)blockIdx.x * blockDim.x + threadIdx.x;
    if (i >= (size_t)B_ROWS * OUT_F) return;
    size_t row = i / OUT_F, col = i % OUT_F;
    float acc = 0.0f;
    const float* xr = x + row * IN_F;
    for (int k = 0; k < IN_F; ++k) acc = fmaf(xr[k], W[(size_t)k * OUT_F + col], acc);
    float un = SIG_TAU * u[i] + acc;
    float s  = (un >= 1.0f) ? 1.0f : 0.0f;
    u_out[i] = un - s;
    s_out[i] = s;
}

// ---------------------------------------------------------------------------
// Kernel 3: eligibility trace (pure bandwidth, float4 vectorized)
// ---------------------------------------------------------------------------
__global__ void ottt_trace(const float* __restrict__ a_hat,
                           const float* __restrict__ x,
                           float* __restrict__ out) {
    size_t i = ((size_t)blockIdx.x * blockDim.x + threadIdx.x) * 4;
    float4 a  = *(const float4*)(a_hat + i);
    float4 xv = *(const float4*)(x + i);
    float4 r;
    r.x = SIG_TAU * a.x + xv.x;
    r.y = SIG_TAU * a.y + xv.y;
    r.z = SIG_TAU * a.z + xv.z;
    r.w = SIG_TAU * a.w + xv.w;
    *(float4*)(out + i) = r;
}

extern "C" void kernel_launch(void* const* d_in, const int* in_sizes, int n_in,
                              void* d_out, int out_size, void* d_ws, size_t ws_size,
                              hipStream_t stream) {
    const float* u     = (const float*)d_in[0];
    const float* a_hat = (const float*)d_in[1];
    const float* x     = (const float*)d_in[2];
    const float* W     = (const float*)d_in[3];

    float* out    = (float*)d_out;
    float* u_out  = out;
    float* ah_out = out + (size_t)B_ROWS * OUT_F;
    float* s_out  = out + 2 * (size_t)B_ROWS * OUT_F;

    if (ws_size >= (size_t)(1024 * 1024)) {
        unsigned char* ws = (unsigned char*)d_ws;
        // Pre-split + pre-swizzle W into bf16 hi/lo B-fragments (1MB of ws).
        ottt_prep_w<<<64, 256, 0, stream>>>(W, ws);
        // Fused split-bf16 WMMA GEMM + LIF.
        ottt_gemm_lif<<<B_ROWS / 32, 256, 0, stream>>>(u, x, ws, u_out, s_out);
    } else {
        ottt_gemm_lif_fallback<<<(B_ROWS * OUT_F + 255) / 256, 256, 0, stream>>>(
            u, x, W, u_out, s_out);
    }
    // Eligibility trace: B*IN/4 float4s.
    ottt_trace<<<(B_ROWS * IN_F / 4) / 256, 256, 0, stream>>>(a_hat, x, ah_out);
}
